// MoELM_24249385353841
// MI455X (gfx1250) — compile-verified, matrix operand order checked by reading, not run on hardware
//
#include <hip/hip_runtime.h>
#include <hip/hip_bf16.h>
#include <math.h>

#define VOCAB 16000
#define BATCH 8
#define TLEN  1024
#define DDIM  64
#define FDIM  256
#define NEXP  8
#define BT    (BATCH*TLEN)   // 8192

typedef __attribute__((ext_vector_type(16))) _Float16 v16h;
typedef __attribute__((ext_vector_type(8)))  _Float16 v8h;
typedef __attribute__((ext_vector_type(8)))  float    v8f;

#define WMMA_F16(a, b, c) \
  __builtin_amdgcn_wmma_f32_16x16x32_f16(false, (a), false, (b), (short)0, (c), false, false)

// ---------------------------------------------------------------------------
// Fragment helpers for v_wmma_f32_16x16x32_f16 (wave32).
// A is 16(M)x32(K) row-major; per ISA layout lane L<16 holds row M=L with
// K = {0..7, 16..23}; lane L>=16 holds row M=L-16 with K = {8..15, 24..31}.
// B fragments are loaded from the *transposed* (N-major) matrix with the same
// pattern, so a single loader serves both.  Rows must be 16B aligned.
__device__ __forceinline__ v16h load_frag(const _Float16* p, int ld, int lane) {
  int r   = lane & 15;
  int off = (lane & 16) >> 1;            // 0 or 8
  const _Float16* q = p + r * ld + off;
  v8h lo = *(const v8h*)(q);             // K = off .. off+7
  v8h hi = *(const v8h*)(q + 16);        // K = 16+off .. 16+off+7
  v16h f;
#pragma unroll
  for (int j = 0; j < 8; ++j) { f[j] = lo[j]; f[j + 8] = hi[j]; }
  return f;
}

// C/D 16x16 f32: VGPR r holds (M=r, N=lane) for lanes 0-15, (M=r+8) for 16-31.
__device__ __forceinline__ void store_tile_f32(float* tile, long ld, int lane, v8f c) {
  int n  = lane & 15;
  int mb = (lane & 16) >> 1;
#pragma unroll
  for (int r = 0; r < 8; ++r) tile[(long)(mb + r) * ld + n] = c[r];
}

// ---------------------------------------------------------------------------
// K0: fp32 -> f16 transpose (dst[c*R + r] = src[r*C + c]), batched over y.
__global__ void moelm_transpose_f16(const float* __restrict__ src,
                                    _Float16* __restrict__ dst, int R, int C) {
  long base = (long)blockIdx.y * R * C;
  int idx = blockIdx.x * blockDim.x + threadIdx.x;
  if (idx < R * C) {
    int r = idx / C, c = idx % C;
    dst[base + (long)c * R + r] = (_Float16)src[base + idx];
  }
}

// K1: embedding gather, fp32 + f16 copies.
__global__ void moelm_gather(const int* __restrict__ x, const float* __restrict__ tab,
                             float* __restrict__ emb, _Float16* __restrict__ embh) {
  int bt = blockIdx.x, d = threadIdx.x;
  float v = tab[(long)x[bt] * DDIM + d];
  emb[(long)bt * DDIM + d]  = v;
  embh[(long)bt * DDIM + d] = (_Float16)v;
}

// K2: XWm = emb @ Wm   (M=8192, N=64, K=64).  4 waves/block, 1 tile each.
__global__ void moelm_xwm(const _Float16* __restrict__ A, const _Float16* __restrict__ WmT,
                          float* __restrict__ C) {
  int mt = blockIdx.x, wave = threadIdx.x >> 5, lane = threadIdx.x & 31;
  const _Float16* a = A + (long)mt * 16 * DDIM;
  const _Float16* b = WmT + wave * 16 * DDIM;  // N-major, ld = K = 64
  v8f c = {};
  v16h a0 = load_frag(a, DDIM, lane), a1 = load_frag(a + 32, DDIM, lane);
  c = WMMA_F16(a0, load_frag(b, DDIM, lane), c);
  c = WMMA_F16(a1, load_frag(b + 32, DDIM, lane), c);
  store_tile_f32(C + (long)mt * 16 * DDIM + wave * 16, DDIM, lane, c);
}

// K3: serial router recurrence + softmax gates.  One block per batch row,
// 64 threads; Um/Wg columns register-resident; mem in LDS (ds_load_b128).
__global__ void moelm_recurrence(const float* __restrict__ XWm, const float* __restrict__ Um,
                                 const float* __restrict__ bm, const float* __restrict__ Wg,
                                 const float* __restrict__ bg, float* __restrict__ gates) {
  const int b = blockIdx.x, d = threadIdx.x;
  __shared__ __align__(16) float mem[DDIM];
  __shared__ float sg[NEXP];
  float um[DDIM];
#pragma unroll
  for (int k = 0; k < DDIM; ++k) um[k] = Um[k * DDIM + d];
  float wg[DDIM];
#pragma unroll
  for (int k = 0; k < DDIM; ++k) wg[k] = (d < NEXP) ? Wg[k * NEXP + d] : 0.f;
  const float bmd = bm[d];
  const float bgv = (d < NEXP) ? bg[d] : 0.f;
  mem[d] = 0.f;
  __syncthreads();
  const float* xrow = XWm + (long)b * TLEN * DDIM + d;
  float xnext = xrow[0];
  for (int t = 0; t < TLEN; ++t) {
    float acc = xnext + bmd;
    if (t + 1 < TLEN) xnext = xrow[(long)(t + 1) * DDIM];   // prefetch (L2-hot)
    const float4* m4 = (const float4*)mem;
#pragma unroll
    for (int k4 = 0; k4 < 16; ++k4) {
      float4 mv = m4[k4];
      acc += mv.x * um[4*k4+0] + mv.y * um[4*k4+1] + mv.z * um[4*k4+2] + mv.w * um[4*k4+3];
    }
    float nm = tanhf(acc);
    __syncthreads();
    mem[d] = nm;
    __syncthreads();
    if (d < NEXP) {
      float l = bgv;
      const float4* mm4 = (const float4*)mem;
#pragma unroll
      for (int k4 = 0; k4 < 16; ++k4) {
        float4 mv = mm4[k4];
        l += mv.x * wg[4*k4+0] + mv.y * wg[4*k4+1] + mv.z * wg[4*k4+2] + mv.w * wg[4*k4+3];
      }
      sg[d] = l;
    }
    __syncthreads();
    if (d < NEXP) {
      float mx = sg[0];
#pragma unroll
      for (int e = 1; e < NEXP; ++e) mx = fmaxf(mx, sg[e]);
      float sum = 0.f;
#pragma unroll
      for (int e = 0; e < NEXP; ++e) sum += __expf(sg[e] - mx);
      gates[((long)b * TLEN + t) * NEXP + d] = __expf(sg[d] - mx) / sum;
    }
  }
}

// K4: fused expert FFN:  h = gelu(X@W1[e]+b1) (WMMA -> LDS f16),
//     eo = h@W2[e]+b2 (WMMA from LDS).  Block = 16 token rows x 1 expert.
#define HLD 272   // 16-byte-aligned LDS row stride for 256-wide h tile
__global__ void moelm_ffn(const _Float16* __restrict__ X, const _Float16* __restrict__ W1T,
                          const float* __restrict__ b1, const _Float16* __restrict__ W2T,
                          const float* __restrict__ b2, float* __restrict__ eo) {
  int mt = blockIdx.x, e = blockIdx.y;
  int wave = threadIdx.x >> 5, lane = threadIdx.x & 31;
  __shared__ __align__(16) _Float16 h[16 * HLD];
  const _Float16* a = X + (long)mt * 16 * DDIM;
  v16h a0 = load_frag(a, DDIM, lane), a1 = load_frag(a + 32, DDIM, lane);
  const _Float16* w1 = W1T + (long)e * FDIM * DDIM;    // [F rows, D] N-major
#pragma unroll
  for (int s = 0; s < 2; ++s) {                        // 8 waves x 2 = 16 n-tiles
    int nt = wave * 2 + s;
    const _Float16* bp = w1 + nt * 16 * DDIM;
    v8f c = {};
    c = WMMA_F16(a0, load_frag(bp, DDIM, lane), c);
    c = WMMA_F16(a1, load_frag(bp + 32, DDIM, lane), c);
    int n = nt * 16 + (lane & 15);
    float bias = b1[e * FDIM + n];
    int mb = (lane & 16) >> 1;
#pragma unroll
    for (int r = 0; r < 8; ++r) {
      float y = c[r] + bias;
      y = 0.5f * y * (1.f + erff(y * 0.70710678118f));  // exact-erf GELU
      h[(mb + r) * HLD + n] = (_Float16)y;
    }
  }
  __syncthreads();
  if (wave < 4) {                                       // 4 d-tiles of 16
    int nt = wave;
    const _Float16* w2 = W2T + (long)e * DDIM * FDIM + nt * 16 * FDIM; // [D rows, F]
    v8f c = {};
#pragma unroll
    for (int kk = 0; kk < 8; ++kk) {                    // K = 256
      v16h af = load_frag(h + kk * 32, HLD, lane);      // from LDS
      v16h bf = load_frag(w2 + kk * 32, FDIM, lane);
      c = WMMA_F16(af, bf, c);
    }
    int dcol = nt * 16 + (lane & 15);
    float bias = b2[e * DDIM + dcol];
    int mb = (lane & 16) >> 1;
    float* o = eo + ((long)e * BT + (long)mt * 16) * DDIM;
#pragma unroll
    for (int r = 0; r < 8; ++r) o[(long)(mb + r) * DDIM + dcol] = c[r] + bias;
  }
}

// K5: gated combine + residual + LayerNorm -> f16 activations for head.
__global__ void moelm_combine_ln(const float* __restrict__ emb, const float* __restrict__ eo,
                                 const float* __restrict__ gates, const float* __restrict__ ln_g,
                                 const float* __restrict__ ln_b, _Float16* __restrict__ outh) {
  int bt = blockIdx.x, d = threadIdx.x;
  float y = emb[(long)bt * DDIM + d];
#pragma unroll
  for (int e = 0; e < NEXP; ++e)
    y += gates[(long)bt * NEXP + e] * eo[((long)e * BT + bt) * DDIM + d];
  __shared__ __align__(16) float r1[DDIM];
  __shared__ __align__(16) float r2[DDIM];
  r1[d] = y;
  __syncthreads();
  float s = 0.f;
  const float4* p4 = (const float4*)r1;
#pragma unroll
  for (int k = 0; k < 16; ++k) { float4 v = p4[k]; s += v.x + v.y + v.z + v.w; }
  float mu = s * (1.f / DDIM);
  float df = y - mu;
  r2[d] = df * df;
  __syncthreads();
  float vv = 0.f;
  const float4* q4 = (const float4*)r2;
#pragma unroll
  for (int k = 0; k < 16; ++k) { float4 v = q4[k]; vv += v.x + v.y + v.z + v.w; }
  vv *= (1.f / DDIM);
  outh[(long)bt * DDIM + d] = (_Float16)(ln_g[d] * df * rsqrtf(vv + 1e-5f) + ln_b[d]);
}

// K6: head GEMM  logits = out @ head_W + head_b  (M=8192, N=16000, K=64).
// Store-bandwidth bound (512 MB fp32).  Block = 16x128 tile, 4 waves; each
// wave keeps A in registers across 2 N-tiles (4 WMMAs), results staged in
// LDS and streamed out as fully-coalesced b128 stores (512 B per row chunk).
__global__ void moelm_head(const _Float16* __restrict__ A, const _Float16* __restrict__ WT,
                           const float* __restrict__ bias, float* __restrict__ out) {
  int mt = blockIdx.x, nb = blockIdx.y;
  int wave = threadIdx.x >> 5, lane = threadIdx.x & 31;
  __shared__ __align__(16) float ot[16 * 128];
  const _Float16* a = A + (long)mt * 16 * DDIM;
  v16h a0 = load_frag(a, DDIM, lane), a1 = load_frag(a + 32, DDIM, lane);
  int n0 = nb * 128;
#pragma unroll
  for (int s = 0; s < 2; ++s) {                         // 2 N-tiles per wave
    int ncol = wave * 32 + s * 16;                      // col offset within block tile
    const _Float16* w = WT + (long)(n0 + ncol) * DDIM;  // N-major head_W^T, ld = 64
    v8f c = {};
    c = WMMA_F16(a0, load_frag(w, DDIM, lane), c);
    c = WMMA_F16(a1, load_frag(w + 32, DDIM, lane), c);
    int n = ncol + (lane & 15);
    float bv = bias[n0 + n];
    int mb = (lane & 16) >> 1;
#pragma unroll
    for (int r = 0; r < 8; ++r) ot[(mb + r) * 128 + n] = c[r] + bv;
  }
  __syncthreads();
  // Stream 16x128 fp32 tile to global: 512 float4's, 128 threads x 4 each.
  const float4* s4 = (const float4*)ot;
  int tid = threadIdx.x;
#pragma unroll
  for (int k = 0; k < 4; ++k) {
    int f   = k * 128 + tid;          // 0..511
    int row = f >> 5;                 // /32 float4 per row
    int c4  = f & 31;
    *(float4*)(out + (long)(mt * 16 + row) * VOCAB + n0 + c4 * 4) = s4[f];
  }
}

// ---------------------------------------------------------------------------
extern "C" void kernel_launch(void* const* d_in, const int* in_sizes, int n_in,
                              void* d_out, int out_size, void* d_ws, size_t ws_size,
                              hipStream_t stream) {
  const int*   x     = (const int*)  d_in[0];
  const float* table = (const float*)d_in[1];
  const float* Wm    = (const float*)d_in[2];
  const float* Um    = (const float*)d_in[3];
  const float* bm    = (const float*)d_in[4];
  const float* Wg    = (const float*)d_in[5];
  const float* bg    = (const float*)d_in[6];
  const float* W1    = (const float*)d_in[7];
  const float* b1    = (const float*)d_in[8];
  const float* W2    = (const float*)d_in[9];
  const float* b2    = (const float*)d_in[10];
  const float* ln_g  = (const float*)d_in[11];
  const float* ln_b  = (const float*)d_in[12];
  const float* headW = (const float*)d_in[13];
  const float* headb = (const float*)d_in[14];
  float* out = (float*)d_out;

  char* ws = (char*)d_ws;
  size_t off = 0;
  auto alloc = [&](size_t bytes) -> void* {
    off = (off + 255) & ~(size_t)255;
    void* p = ws + off;
    off += bytes;
    return p;
  };
  float*    emb   = (float*)   alloc((size_t)BT * DDIM * 4);
  _Float16* embh  = (_Float16*)alloc((size_t)BT * DDIM * 2);
  _Float16* WmT   = (_Float16*)alloc((size_t)DDIM * DDIM * 2);
  _Float16* W1T   = (_Float16*)alloc((size_t)NEXP * DDIM * FDIM * 2);
  _Float16* W2T   = (_Float16*)alloc((size_t)NEXP * FDIM * DDIM * 2);
  _Float16* hWT   = (_Float16*)alloc((size_t)VOCAB * DDIM * 2);
  float*    XWm   = (float*)   alloc((size_t)BT * DDIM * 4);
  float*    gates = (float*)   alloc((size_t)BT * NEXP * 4);
  float*    eo    = (float*)   alloc((size_t)NEXP * BT * DDIM * 4);
  _Float16* outh  = (_Float16*)alloc((size_t)BT * DDIM * 2);
  (void)ws_size; (void)in_sizes; (void)n_in; (void)out_size;

  // Weight conversion/transposition to N-major f16.
  moelm_transpose_f16<<<dim3((DDIM*DDIM + 255)/256, 1),    256, 0, stream>>>(Wm,    WmT, DDIM, DDIM);
  moelm_transpose_f16<<<dim3((DDIM*FDIM + 255)/256, NEXP), 256, 0, stream>>>(W1,    W1T, DDIM, FDIM);
  moelm_transpose_f16<<<dim3((FDIM*DDIM + 255)/256, NEXP), 256, 0, stream>>>(W2,    W2T, FDIM, DDIM);
  moelm_transpose_f16<<<dim3((DDIM*VOCAB + 255)/256, 1),   256, 0, stream>>>(headW, hWT, DDIM, VOCAB);

  moelm_gather<<<BT, DDIM, 0, stream>>>(x, table, emb, embh);
  moelm_xwm<<<BT / 16, 128, 0, stream>>>(embh, WmT, XWm);
  moelm_recurrence<<<BATCH, DDIM, 0, stream>>>(XWm, Um, bm, Wg, bg, gates);
  moelm_ffn<<<dim3(BT / 16, NEXP), 256, 0, stream>>>(embh, W1T, b1, W2T, b2, eo);
  moelm_combine_ln<<<BT, DDIM, 0, stream>>>(emb, eo, gates, ln_g, ln_b, outh);
  moelm_head<<<dim3(BT / 16, VOCAB / 128), 128, 0, stream>>>(outh, hWT, headb, out);
}
